// LiDARPriorQueryGenerator_81071802679547
// MI455X (gfx1250) — matrix operand order repaired and checked by reading, not compiled, not run on hardware
//
#include <hip/hip_runtime.h>
#include <hip/hip_bf16.h>

typedef __attribute__((ext_vector_type(16))) _Float16 v16h;
typedef __attribute__((ext_vector_type(8)))  _Float16 v8h;
typedef __attribute__((ext_vector_type(8)))  float    v8f;
typedef __attribute__((ext_vector_type(4)))  unsigned int u32x4;
typedef __attribute__((ext_vector_type(8)))  int i32x8;
typedef __attribute__((ext_vector_type(4)))  int i32x4;

union HFrag { v16h v; v8h h[2]; };

#define HWD   128
#define CCH   256
#define NPOS  16384   // 128*128
#define NB    4
#define KSEL  100

// ---------------------------------------------------------------------------
// Prep kernels
// ---------------------------------------------------------------------------

// NCHW fp32 -> NHWC f16   (4*128*128*256 elements)
__global__ __launch_bounds__(256) void cvt_nhwc_f16(
    const float* __restrict__ x, _Float16* __restrict__ out) {
  size_t idx = (size_t)blockIdx.x * 256 + threadIdx.x;
  int c  = (int)(idx & 255);
  int xx = (int)((idx >> 8) & 127);
  int yy = (int)((idx >> 15) & 127);
  int b  = (int)(idx >> 22);
  out[idx] = (_Float16)x[(((size_t)(b * CCH + c) * HWD + yy) * HWD + xx)];
}

// conv weight (O,I,3,3) fp32 -> fragment-linear [kblk][oc][32] f16
__global__ __launch_bounds__(256) void repack_w(
    const float* __restrict__ w, _Float16* __restrict__ wf) {
  int idx = blockIdx.x * 256 + threadIdx.x;         // 72*256*32 = 589824
  int j  = idx & 31;
  int oc = (idx >> 5) & 255;
  int kb = idx >> 13;
  int ic = ((kb & 7) << 5) + j;
  int kk = kb >> 3;
  int ky = kk / 3, kx = kk % 3;
  wf[idx] = (_Float16)w[(((size_t)(oc * CCH + ic) * 3 + ky) * 3 + kx)];
}

__global__ __launch_bounds__(256) void bn_prep(
    const float* g1, const float* b1, const float* m1, const float* v1,
    const float* g2, const float* b2, const float* m2, const float* v2,
    float* s1, float* sh1, float* s2, float* sh2, _Float16* zbuf) {
  int c = threadIdx.x;
  if (c < CCH) {
    float a = g1[c] * rsqrtf(v1[c] + 1e-5f);
    s1[c] = a; sh1[c] = b1[c] - m1[c] * a;
    float d = g2[c] * rsqrtf(v2[c] + 1e-5f);
    s2[c] = d; sh2[c] = b2[c] - m2[c] * d;
    zbuf[c] = (_Float16)0.0f;       // 512B zero page for halo lanes
  }
}

__global__ __launch_bounds__(256) void init_scores(
    float* __restrict__ s, const float* __restrict__ bias) {
  int idx = blockIdx.x * 256 + threadIdx.x;         // 65536
  s[idx] = bias[0];
}

// ---------------------------------------------------------------------------
// Tensor Data Mover: stage one 4x(64x32) f16 weight chunk (16KB) into LDS.
// D# per ISA 8.3/8.4: 2-D tile, data_size=8B (code 3), row = 4KB (512 x 8B),
// 4 rows, row stride 16KB (2048 x 8B). group0.type=2 ("image").
// ---------------------------------------------------------------------------
__device__ __forceinline__ void tdm_load_b(const _Float16* gsrc, unsigned lds_off) {
  unsigned long long ga = (unsigned long long)(uintptr_t)gsrc;
  u32x4 g0 = { 1u,                                   // count=1, is_restore=0
               lds_off,                              // lds_addr (bytes)
               (unsigned)(ga & 0xFFFFFFFFu),         // global_addr[31:0]
               ((unsigned)((ga >> 32) & 0x01FFFFFFu)) | 0x80000000u }; // [56:32] | type=2
  i32x8 g1 = { (int)0x30000u,        // data_size=3 (8B units)
               (int)(512u << 16),    // tensor_dim0[15:0] in [31:16]
               (int)(4u << 16),      // tensor_dim0 hi=0 | tensor_dim1[15:0]
               (int)(512u << 16),    // tensor_dim1 hi=0 | tile_dim0=512
               4,                    // tile_dim1=4, tile_dim2=0
               2048,                 // tensor_dim0_stride lo32 (8B units)
               0, 0 };
  i32x4 z4 = { 0, 0, 0, 0 };
#if defined(__clang_major__) && (__clang_major__ >= 23)
  i32x8 z8 = { 0, 0, 0, 0, 0, 0, 0, 0 };
  __builtin_amdgcn_tensor_load_to_lds(g0, g1, z4, z4, z8, 0);
#else
  __builtin_amdgcn_tensor_load_to_lds(g0, g1, z4, z4, 0);
#endif
}

// ---------------------------------------------------------------------------
// Implicit-GEMM 3x3 conv core. 18 chunks of 4 k-blocks (K=32 each).
// B double-buffered in LDS via TDM; A + B fragments register double-buffered
// so chunk-(c+1) DMA and step-(t+1) loads overlap step-t WMMAs.
// ---------------------------------------------------------------------------
__device__ __forceinline__ void conv_core(
    const _Float16* __restrict__ xin, const _Float16* __restrict__ wfrag,
    const _Float16* __restrict__ zbuf, _Float16 (&bstage)[2][8192],
    int b, int y, int xbase, int n0, int wave, int lane, v8f acc[4]) {
  const int m  = lane & 15;
  const int kg = lane >> 4;
  const int xm = xbase + m;
  if (wave == 0)
    tdm_load_b(wfrag + (size_t)n0 * 32, (unsigned)(size_t)(void*)&bstage[0][0]);
  #pragma unroll 1
  for (int c = 0; c < 18; ++c) {
    if (wave == 0) __builtin_amdgcn_s_wait_tensorcnt(0);
    __syncthreads();                     // chunk c staged; chunk c-1 fully consumed
    if (wave == 0 && c < 17)
      tdm_load_b(wfrag + (size_t)(c + 1) * 32768 + (size_t)n0 * 32,
                 (unsigned)(size_t)(void*)&bstage[(c + 1) & 1][0]);
    const int kk = c >> 1;               // filter tap 0..8
    const int dy = kk / 3 - 1, dx = kk % 3 - 1;
    const int yy = y + dy, xx = xm + dx;
    const bool valid = ((unsigned)yy < (unsigned)HWD) & ((unsigned)xx < (unsigned)HWD);
    const _Float16* ap = valid
        ? xin + (((size_t)((b * HWD + yy) * HWD + xx)) << 8) + ((c & 1) << 7) + (kg << 3)
        : zbuf;                          // zero page: unconditional loads, no EXEC games
    const _Float16* lb = &bstage[c & 1][m * 32 + (kg << 3)];
    HFrag a[2]; HFrag bf[2][4];
    a[0].h[0] = *(const v8h*)(ap);
    a[0].h[1] = *(const v8h*)(ap + 16);
    #pragma unroll
    for (int j = 0; j < 4; ++j) {
      bf[0][j].h[0] = *(const v8h*)(lb + j * 512);
      bf[0][j].h[1] = *(const v8h*)(lb + j * 512 + 16);
    }
    #pragma unroll
    for (int t = 0; t < 4; ++t) {
      const int cur = t & 1, nxt = cur ^ 1;
      if (t < 3) {                       // prefetch step t+1 into the other slot
        const _Float16* ap2 = ap + (t + 1) * 32;
        a[nxt].h[0] = *(const v8h*)(ap2);
        a[nxt].h[1] = *(const v8h*)(ap2 + 16);
        const _Float16* lb2 = lb + (t + 1) * 2048;
        #pragma unroll
        for (int j = 0; j < 4; ++j) {
          bf[nxt][j].h[0] = *(const v8h*)(lb2 + j * 512);
          bf[nxt][j].h[1] = *(const v8h*)(lb2 + j * 512 + 16);
        }
      }
      #pragma unroll
      for (int j = 0; j < 4; ++j)
        acc[j] = __builtin_amdgcn_wmma_f32_16x16x32_f16(
            false, a[cur].v, false, bf[cur][j].v, (short)0, acc[j], false, false);
    }
  }
}

// Conv1: BN + ReLU fused, write feat fp32 (gather path) + f16 (conv2 input)
__global__ __launch_bounds__(128) void conv1_wmma(
    const _Float16* __restrict__ xin, const _Float16* __restrict__ wf,
    const _Float16* __restrict__ zbuf,
    const float* __restrict__ scale, const float* __restrict__ shift,
    float* __restrict__ out32, _Float16* __restrict__ out16) {
  __shared__ _Float16 bstage[2][8192];   // 2 x 16KB
  const int wg = blockIdx.x;
  const int n0 = (wg & 3) << 6;
  const int mt = (wg >> 2) & 255;
  const int b  = wg >> 10;
  const int p0 = mt << 6;
  const int y  = p0 >> 7;
  const int x0 = p0 & 127;
  const int w    = threadIdx.x >> 5;
  const int lane = threadIdx.x & 31;
  v8f acc[4] = {};
  conv_core(xin, wf, zbuf, bstage, b, y, x0 + (w << 4), n0, w, lane, acc);
  const int mhalf = (lane >> 4) << 3;
  const int n = lane & 15;
  #pragma unroll
  for (int j = 0; j < 4; ++j) {
    const int oc = n0 + (j << 4) + n;
    const float sc = scale[oc], sh = shift[oc];
    #pragma unroll
    for (int r = 0; r < 8; ++r) {
      const int p = p0 + (w << 4) + mhalf + r;
      float v = fmaxf(acc[j][r] * sc + sh, 0.0f);
      const size_t o = (((size_t)b * NPOS + p) << 8) + oc;
      out32[o] = v;
      out16[o] = (_Float16)v;
    }
  }
}

// Conv2: BN + ReLU + 1x1 heatmap dot fused; never materializes h.
__global__ __launch_bounds__(128) void conv2_wmma(
    const _Float16* __restrict__ xin, const _Float16* __restrict__ wf,
    const _Float16* __restrict__ zbuf,
    const float* __restrict__ scale, const float* __restrict__ shift,
    const float* __restrict__ w2, float* __restrict__ scores) {
  __shared__ _Float16 bstage[2][8192];
  __shared__ float sred[64];
  const int wg = blockIdx.x;
  const int n0 = (wg & 3) << 6;
  const int mt = (wg >> 2) & 255;
  const int b  = wg >> 10;
  const int p0 = mt << 6;
  const int y  = p0 >> 7;
  const int x0 = p0 & 127;
  const int w    = threadIdx.x >> 5;
  const int lane = threadIdx.x & 31;
  if (threadIdx.x < 64) sred[threadIdx.x] = 0.0f;
  v8f acc[4] = {};
  conv_core(xin, wf, zbuf, bstage, b, y, x0 + (w << 4), n0, w, lane, acc);
  const int mhalf = (lane >> 4) << 3;
  const int n = lane & 15;
  float pl[8] = {0.f, 0.f, 0.f, 0.f, 0.f, 0.f, 0.f, 0.f};
  #pragma unroll
  for (int j = 0; j < 4; ++j) {
    const int oc = n0 + (j << 4) + n;
    const float sc = scale[oc], sh = shift[oc], ww = w2[oc];
    #pragma unroll
    for (int r = 0; r < 8; ++r) {
      float v = fmaxf(acc[j][r] * sc + sh, 0.0f);
      pl[r] += v * ww;
    }
  }
  #pragma unroll
  for (int r = 0; r < 8; ++r)
    atomicAdd(&sred[(w << 4) + mhalf + r], pl[r]);   // ds_add_f32
  __syncthreads();
  if (threadIdx.x < 64)
    unsafeAtomicAdd(&scores[b * NPOS + p0 + threadIdx.x], sred[threadIdx.x]);
}

// ---------------------------------------------------------------------------
// Top-K = 100 per batch. Scores live in registers (64/thread, unrolled),
// claimed entries masked via a 64-bit register mask. Sigmoid only on winners.
// ---------------------------------------------------------------------------
__global__ __launch_bounds__(256) void topk_kernel(
    const float* __restrict__ scores, int* __restrict__ inds,
    float* __restrict__ out_ts) {
  __shared__ float bv[256];
  __shared__ int   bi[256];
  const int b = blockIdx.x, t = threadIdx.x;
  float loc[64];
  #pragma unroll
  for (int i = 0; i < 64; ++i) loc[i] = scores[b * NPOS + i * 256 + t];
  unsigned long long mask = 0ull;
  for (int k = 0; k < KSEL; ++k) {
    float best = -3.4e38f; int bidx = 1 << 30;
    #pragma unroll
    for (int i = 0; i < 64; ++i) {
      float v = ((mask >> i) & 1ull) ? -3.4e38f : loc[i];
      int gi = i * 256 + t;
      if (v > best || (v == best && gi < bidx)) { best = v; bidx = gi; }
    }
    bv[t] = best; bi[t] = bidx;
    __syncthreads();
    for (int s = 128; s > 0; s >>= 1) {
      if (t < s) {
        float v2 = bv[t + s]; int i2 = bi[t + s];
        if (v2 > bv[t] || (v2 == bv[t] && i2 < bi[t])) { bv[t] = v2; bi[t] = i2; }
      }
      __syncthreads();
    }
    int gi = bi[0]; float gv = bv[0];
    if (t == 0) {
      inds[b * KSEL + k] = gi;
      out_ts[b * KSEL + k] = 1.0f / (1.0f + __expf(-gv));
    }
    if ((gi & 255) == t) mask |= 1ull << (gi >> 8);
    __syncthreads();
  }
}

// ---------------------------------------------------------------------------
// Per-prior head: gather -> FC1 -> LayerNorm -> ReLU -> FC2 -> *(1+score),
// plus 8 small head dots and anchor refinement. One WG per (b,k).
// ---------------------------------------------------------------------------
__global__ __launch_bounds__(256) void prior_kernel(
    const float* __restrict__ feat, const int* __restrict__ inds,
    const float* __restrict__ ts,
    const float* __restrict__ fw1, const float* __restrict__ fb1,
    const float* __restrict__ lng, const float* __restrict__ lnb,
    const float* __restrict__ fw2, const float* __restrict__ fb2,
    const float* __restrict__ zw, const float* __restrict__ zb,
    const float* __restrict__ dw, const float* __restrict__ db,
    const float* __restrict__ yw, const float* __restrict__ yb,
    const float* __restrict__ vw, const float* __restrict__ vb,
    const float* __restrict__ anchor,
    float* __restrict__ out_pf, float* __restrict__ out_pa) {
  __shared__ float g[256], xr[256], red[256], hv[8];
  const int bk = blockIdx.x;
  const int b = bk / KSEL, k = bk % KSEL;
  const int t = threadIdx.x;
  const int p = inds[bk];
  g[t] = feat[(((size_t)b * NPOS + p) << 8) + t];
  __syncthreads();

  float acc = fb1[t];
  const float* wr = fw1 + (size_t)t * 256;
  #pragma unroll 8
  for (int i = 0; i < 256; ++i) acc += wr[i] * g[i];

  red[t] = acc;
  __syncthreads();
  for (int s = 128; s > 0; s >>= 1) {
    if (t < s) red[t] += red[t + s];
    __syncthreads();
  }
  float mu = red[0] * (1.0f / 256.0f);
  __syncthreads();
  float d = acc - mu;
  red[t] = d * d;
  __syncthreads();
  for (int s = 128; s > 0; s >>= 1) {
    if (t < s) red[t] += red[t + s];
    __syncthreads();
  }
  float var = red[0] * (1.0f / 256.0f);
  float xn = d * rsqrtf(var + 1e-5f) * lng[t] + lnb[t];
  xn = fmaxf(xn, 0.0f);
  xr[t] = xn;
  __syncthreads();

  float a2 = fb2[t];
  const float* w2r = fw2 + (size_t)t * 256;
  #pragma unroll 8
  for (int i = 0; i < 256; ++i) a2 += w2r[i] * xr[i];
  float sc = ts[bk];
  out_pf[(size_t)bk * 256 + t] = a2 * (1.0f + sc);

  if (t < 8) {
    const float* hw; float hb;
    if      (t == 0) { hw = zw;                hb = zb[0];     }
    else if (t <= 3) { hw = dw + (t - 1) * 256; hb = db[t - 1]; }
    else if (t <= 5) { hw = yw + (t - 4) * 256; hb = yb[t - 4]; }
    else             { hw = vw + (t - 6) * 256; hb = vb[t - 6]; }
    float ha = hb;
    for (int i = 0; i < 256; ++i) ha += hw[i] * g[i];
    hv[t] = ha;
  }
  __syncthreads();

  if (t == 0) {
    const int xs = p & 127, ys = p >> 7;
    const float* pa = anchor + ((size_t)b * 900 + k) * 11;
    float o[11];
    o[0] = ((float)xs + 0.5f) * 0.8f - 51.2f;
    o[1] = ((float)ys + 0.5f) * 0.8f - 51.2f;
    o[2] = pa[2] + 0.5f * hv[0];
    #pragma unroll
    for (int j = 0; j < 3; ++j)
      o[3 + j] = pa[3 + j] + 0.2f * fminf(fmaxf(hv[1 + j], -1.0f), 1.0f);
    float y0 = tanhf(hv[4]), y1 = tanhf(hv[5]);
    float nrm = fmaxf(sqrtf(y0 * y0 + y1 * y1), 1e-6f);
    o[6] = 0.7f * pa[6] + 0.3f * (y0 / nrm);
    o[7] = 0.7f * pa[7] + 0.3f * (y1 / nrm);
    o[8] = pa[8] + 0.2f * fminf(fmaxf(hv[6], -2.0f), 2.0f);
    o[9] = pa[9] + 0.2f * fminf(fmaxf(hv[7], -2.0f), 2.0f);
    o[10] = pa[10];
    #pragma unroll
    for (int j = 0; j < 11; ++j) out_pa[(size_t)bk * 11 + j] = o[j];
  }
}

// ---------------------------------------------------------------------------
// Host launcher
// ---------------------------------------------------------------------------
extern "C" void kernel_launch(void* const* d_in, const int* in_sizes, int n_in,
                              void* d_out, int out_size, void* d_ws, size_t ws_size,
                              hipStream_t stream) {
  const float* bev     = (const float*)d_in[0];
  const float* anchor  = (const float*)d_in[1];
  const float* sharedw = (const float*)d_in[2];
  const float* bn1_g = (const float*)d_in[3];
  const float* bn1_b = (const float*)d_in[4];
  const float* bn1_m = (const float*)d_in[5];
  const float* bn1_v = (const float*)d_in[6];
  const float* objw1 = (const float*)d_in[7];
  const float* bn2_g = (const float*)d_in[8];
  const float* bn2_b = (const float*)d_in[9];
  const float* bn2_m = (const float*)d_in[10];
  const float* bn2_v = (const float*)d_in[11];
  const float* objw2 = (const float*)d_in[12];
  const float* objb2 = (const float*)d_in[13];
  const float* fpw1  = (const float*)d_in[14];
  const float* fpb1  = (const float*)d_in[15];
  const float* lng   = (const float*)d_in[16];
  const float* lnb   = (const float*)d_in[17];
  const float* fpw2  = (const float*)d_in[18];
  const float* fpb2  = (const float*)d_in[19];
  const float* zw = (const float*)d_in[20];
  const float* zb = (const float*)d_in[21];
  const float* dw = (const float*)d_in[22];
  const float* db = (const float*)d_in[23];
  const float* yw = (const float*)d_in[24];
  const float* yb = (const float*)d_in[25];
  const float* vw = (const float*)d_in[26];
  const float* vb = (const float*)d_in[27];

  // workspace carve-up (~137 MB total)
  char* ws = (char*)d_ws;
  _Float16* xbf    = (_Float16*)ws;  ws += (size_t)NB * NPOS * CCH * 2;   // 32 MB
  _Float16* wf1    = (_Float16*)ws;  ws += (size_t)72 * CCH * 32 * 2;     // 1.125 MB
  _Float16* wf2    = (_Float16*)ws;  ws += (size_t)72 * CCH * 32 * 2;
  float* s1        = (float*)ws;     ws += CCH * 4;
  float* sh1       = (float*)ws;     ws += CCH * 4;
  float* s2        = (float*)ws;     ws += CCH * 4;
  float* sh2       = (float*)ws;     ws += CCH * 4;
  _Float16* zbuf   = (_Float16*)ws;  ws += CCH * 2;                       // zero page
  float* feat32    = (float*)ws;     ws += (size_t)NB * NPOS * CCH * 4;   // 64 MB
  _Float16* feat16 = (_Float16*)ws;  ws += (size_t)NB * NPOS * CCH * 2;   // 32 MB
  float* scores    = (float*)ws;     ws += (size_t)NB * NPOS * 4;         // 256 KB
  int* inds        = (int*)ws;       ws += (size_t)NB * KSEL * 4;

  float* out    = (float*)d_out;
  float* out_pf = out;                       // 4*100*256
  float* out_pa = out + NB * KSEL * 256;     // 4*100*11
  float* out_ts = out_pa + NB * KSEL * 11;   // 4*100

  cvt_nhwc_f16 <<<(NB * NPOS * CCH) / 256, 256, 0, stream>>>(bev, xbf);
  repack_w     <<<(72 * CCH * 32) / 256, 256, 0, stream>>>(sharedw, wf1);
  repack_w     <<<(72 * CCH * 32) / 256, 256, 0, stream>>>(objw1, wf2);
  bn_prep      <<<1, 256, 0, stream>>>(bn1_g, bn1_b, bn1_m, bn1_v,
                                       bn2_g, bn2_b, bn2_m, bn2_v,
                                       s1, sh1, s2, sh2, zbuf);
  init_scores  <<<(NB * NPOS) / 256, 256, 0, stream>>>(scores, objb2);

  const int conv_grid = NB * (NPOS / 64) * (CCH / 64);   // 4096
  conv1_wmma <<<conv_grid, 128, 0, stream>>>(xbf, wf1, zbuf, s1, sh1, feat32, feat16);
  conv2_wmma <<<conv_grid, 128, 0, stream>>>(feat16, wf2, zbuf, s2, sh2, objw2, scores);

  topk_kernel  <<<NB, 256, 0, stream>>>(scores, inds, out_ts);
  prior_kernel <<<NB * KSEL, 256, 0, stream>>>(feat32, inds, out_ts,
                                               fpw1, fpb1, lng, lnb, fpw2, fpb2,
                                               zw, zb, dw, db, yw, yb, vw, vb,
                                               anchor, out_pf, out_pa);
}
